// AttentionBase_10385230922227
// MI455X (gfx1250) — matrix operand order repaired and checked by reading, not compiled, not run on hardware
//
#include <hip/hip_runtime.h>
#include <math.h>

#define HEADS 16
#define CDIM  128
#define EDIM  2048
#define BATCH 2
#define SEQ   2048

typedef __attribute__((ext_vector_type(16))) __bf16 v16bf;
typedef __attribute__((ext_vector_type(8)))  float  v8f;
typedef __attribute__((ext_vector_type(2)))  float  v2f;
typedef __attribute__((ext_vector_type(2)))  __bf16 v2bf;
typedef __attribute__((ext_vector_type(4)))  int    v4i;

#define AS1 __attribute__((address_space(1)))
#define AS3 __attribute__((address_space(3)))

// bf16 fragment viewed as 8 packed dwords / 2 uint4
typedef union { v16bf bf; unsigned u[8]; uint4 q[2]; } Frag;

// pack two f32 into one bf16x2 dword (lowers to v_cvt_pk_bf16_f32)
static __device__ __forceinline__ unsigned pk_bf16(float x, float y) {
    v2f f; f.x = x; f.y = y;
    v2bf h = __builtin_convertvector(f, v2bf);
    union { v2bf v; unsigned u; } c; c.v = h;
    return c.u;
}

#if __has_builtin(__builtin_amdgcn_global_load_async_to_lds_b128) && \
    __has_builtin(__builtin_amdgcn_s_wait_asynccnt)
#define HAVE_ASYNC 1
#endif

// 16-byte global->LDS copy (async when available)
static __device__ __forceinline__ void copy16(const void* g, void* l) {
#ifdef HAVE_ASYNC
    __builtin_amdgcn_global_load_async_to_lds_b128((AS1 v4i*)g, (AS3 v4i*)l, 0, 0);
#else
    *(uint4*)l = *(const uint4*)g;
#endif
}
static __device__ __forceinline__ void copy_wait() {
#ifdef HAVE_ASYNC
    __builtin_amdgcn_s_wait_asynccnt(0);
#endif
}

// ---------------------------------------------------------------------------
// Pre-pass 1: keys f32 [b,s,h,c] -> Kpack bf16 [(b,h), s, c] (pure convert)
// ---------------------------------------------------------------------------
__global__ __launch_bounds__(256) void pack_k_kernel(
    const float* __restrict__ keys, unsigned* __restrict__ kp)
{
    const int i = blockIdx.x * 256 + threadIdx.x;      // dword index
    const int cpair = i & 63;
    const int rest  = i >> 6;
    const int s     = rest & (SEQ - 1);
    const int bh    = rest >> 11;
    const int b     = bh >> 4, h = bh & 15;
    float2 f = *(const float2*)(keys + ((size_t)(b * SEQ + s) * EDIM + h * CDIM + cpair * 2));
    kp[i] = pk_bf16(f.x, f.y);
}

// ---------------------------------------------------------------------------
// Pre-pass 2: values f32 [b,s,h,d] -> Vpack bf16 [(b,h), d, s] (transpose)
// One 256-thread block per (b,h,32-key tile); thread owns (d, 16 keys).
// ---------------------------------------------------------------------------
__global__ __launch_bounds__(256) void pack_v_kernel(
    const float* __restrict__ values, unsigned short* __restrict__ vp)
{
    const int nkt = SEQ / 32;
    const int kt  = (blockIdx.x % nkt) * 32;
    const int bh  = blockIdx.x / nkt;
    const int b   = bh >> 4, h = bh & 15;
    const int sd  = threadIdx.x >> 1;          // 0..127 head dim
    const int skh = (threadIdx.x & 1) * 16;    // 0 | 16 keys
    const float* gv = values + ((size_t)(b * SEQ + kt + skh) * EDIM + h * CDIM + sd);
    float f[16];
    #pragma unroll
    for (int i = 0; i < 16; ++i) f[i] = gv[(size_t)i * EDIM];
    uint4 t0, t1;
    t0.x = pk_bf16(f[0],  f[1]);  t0.y = pk_bf16(f[2],  f[3]);
    t0.z = pk_bf16(f[4],  f[5]);  t0.w = pk_bf16(f[6],  f[7]);
    t1.x = pk_bf16(f[8],  f[9]);  t1.y = pk_bf16(f[10], f[11]);
    t1.z = pk_bf16(f[12], f[13]); t1.w = pk_bf16(f[14], f[15]);
    unsigned short* dst = vp + ((size_t)(bh * CDIM + sd) * SEQ + kt + skh);
    *(uint4*)(dst)     = t0;
    *(uint4*)(dst + 8) = t1;
}

// ---------------------------------------------------------------------------
// Pre-pass 3: w_out f32 [j,e] -> Wpack bf16 [j,e]
// ---------------------------------------------------------------------------
__global__ __launch_bounds__(256) void pack_w_kernel(
    const float* __restrict__ w, unsigned* __restrict__ wp)
{
    const int i = blockIdx.x * 256 + threadIdx.x;      // dword index
    const int epair = i & 1023;
    const int j     = i >> 10;
    float2 f = *(const float2*)(w + ((size_t)j * EDIM + epair * 2));
    wp[i] = pk_bf16(f.x, f.y);
}

// ---------------------------------------------------------------------------
// Fused flash attention. Block = 8 waves = one (b, head, 128-query tile).
// K/V tiles are pre-converted bf16 in exactly the LDS layout, so staging is
// a pure async global->LDS copy; fragment reads are contiguous ds_load_b128.
// S^T = K*Q^T keeps softmax per-lane; P repacks in-lane. Output stored bf16.
// ---------------------------------------------------------------------------
__global__ __launch_bounds__(256) void attn_fwd_kernel(
    const unsigned short* __restrict__ kp, const unsigned short* __restrict__ vp,
    const float* __restrict__ queries, const unsigned char* __restrict__ mask,
    unsigned short* __restrict__ attn_out)
{
    __shared__ __align__(16) unsigned short lK[32][128]; // [key][c]  8 KB
    __shared__ __align__(16) unsigned short lV[128][32]; // [d][key]  8 KB

    const int tid  = threadIdx.x;
    const int lane = tid & 31;
    const int wv   = tid >> 5;
    const int col  = lane & 15;
    const int hlf  = lane >> 4;

    const int nq   = SEQ / 128;
    const int qs   = blockIdx.x % nq;
    const int bh   = blockIdx.x / nq;
    const int head = bh & 15;
    const int b    = bh >> 4;
    const int qbase = qs * 128 + wv * 16;
    const float scale = 0.08838834764831845f; // 1/sqrt(128)
    const size_t baseBS = (size_t)b * SEQ;

    // per-wave Q^T B-fragments (scaled, converted once)
    Frag qb[4];
    #pragma unroll
    for (int cc = 0; cc < 4; ++cc) {
        const float* qp = queries +
            ((baseBS + qbase + col) * EDIM + head * CDIM + cc * 32 + hlf * 16);
        #pragma unroll
        for (int v = 0; v < 8; ++v) {
            float2 f = *(const float2*)(qp + 2 * v);
            qb[cc].u[v] = pk_bf16(f.x * scale, f.y * scale);
        }
    }

    v8f o[8];
    #pragma unroll
    for (int dc = 0; dc < 8; ++dc) o[dc] = v8f{};
    float m = -1e30f, l = 0.0f;

    // staging addresses (32 bytes per thread per matrix)
    const char* gK = (const char*)(kp + (size_t)bh * SEQ * CDIM) + tid * 32;
    const int   sd  = tid >> 1;
    const int   skh = (tid & 1) * 16;
    const char* gV = (const char*)(vp + ((size_t)bh * CDIM + sd) * SEQ + skh);
    char* lKp = (char*)&lK[0][0] + tid * 32;
    char* lVp = (char*)&lV[sd][skh];

    for (int kt = 0; kt < SEQ; kt += 32) {
        // ---- stage K/V tile (pure copy; async when available) ----
        copy16(gK + (size_t)kt * 256,      lKp);
        copy16(gK + (size_t)kt * 256 + 16, lKp + 16);
        copy16(gV + (size_t)kt * 2,        lVp);
        copy16(gV + (size_t)kt * 2 + 16,   lVp + 16);
        copy_wait();
        __syncthreads();

        // ---- scores S^T = K * Q^T ----
        v8f st[2];
        #pragma unroll
        for (int sub = 0; sub < 2; ++sub) {
            v8f acc = v8f{};
            #pragma unroll
            for (int cc = 0; cc < 4; ++cc) {
                Frag ka;  // A: lane row = key, K = 16*hlf + 2v + e (contig)
                const unsigned short* kr = &lK[sub * 16 + col][cc * 32 + 16 * hlf];
                ka.q[0] = *(const uint4*)(kr);
                ka.q[1] = *(const uint4*)(kr + 8);
                acc = __builtin_amdgcn_wmma_f32_16x16x32_bf16(
                    false, ka.bf, false, qb[cc].bf, (short)0, acc, false, false);
            }
            st[sub] = acc;
        }

        // ---- online softmax (row = query = lane%16) ----
        float p0[8], p1[8], tmax = -1e30f;
        #pragma unroll
        for (int r = 0; r < 8; ++r) {
            float s0 = mask[baseBS + kt + r + hlf * 8]      ? st[0][r] : -1e30f;
            float s1 = mask[baseBS + kt + 16 + r + hlf * 8] ? st[1][r] : -1e30f;
            p0[r] = s0; p1[r] = s1;
            tmax = fmaxf(tmax, fmaxf(s0, s1));
        }
        tmax = fmaxf(tmax, __shfl_xor(tmax, 16, 32));
        const float newm  = fmaxf(m, tmax);
        const float alpha = __expf(m - newm);
        float rsum = 0.0f;
        #pragma unroll
        for (int r = 0; r < 8; ++r) {
            p0[r] = __expf(p0[r] - newm);
            p1[r] = __expf(p1[r] - newm);
            rsum += p0[r] + p1[r];
        }
        rsum += __shfl_xor(rsum, 16, 32);
        l = l * alpha + rsum;
        m = newm;
        #pragma unroll
        for (int r = 0; r < 8; ++r) {
            const float ar = __shfl(alpha, hlf * 8 + r, 32);
            #pragma unroll
            for (int dc = 0; dc < 8; ++dc) o[dc][r] *= ar;
        }

        // in-lane repack of P (16q x 32k) into A layout
        Frag pa;
        #pragma unroll
        for (int v = 0; v < 4; ++v) pa.u[v] = pk_bf16(p0[2 * v], p0[2 * v + 1]);
        #pragma unroll
        for (int v = 4; v < 8; ++v) pa.u[v] = pk_bf16(p1[2 * (v - 4)], p1[2 * (v - 4) + 1]);

        // ---- PV: O += P * V ----
        #pragma unroll
        for (int dc = 0; dc < 8; ++dc) {
            Frag vb;  // B: lane col = d, K(key) = (v/4)*16 + (v%4)*2 + e + 8*hlf
            const unsigned short* vr = &lV[dc * 16 + col][hlf * 8];
            vb.q[0] = *(const uint4*)(vr);
            vb.q[1] = *(const uint4*)(vr + 16);
            o[dc] = __builtin_amdgcn_wmma_f32_16x16x32_bf16(
                false, pa.bf, false, vb.bf, (short)0, o[dc], false, false);
        }
        __syncthreads();
    }

    // ---- normalize + store bf16 [b, q, head, d] ----
    const float linv = 1.0f / l;
    #pragma unroll
    for (int r = 0; r < 8; ++r) {
        const float lr = __shfl(linv, hlf * 8 + r, 32);
        const int q = qbase + r + hlf * 8;
        unsigned short* op = attn_out + ((baseBS + q) * EDIM + head * CDIM + col);
        #pragma unroll
        for (int dc = 0; dc < 8; ++dc)
            op[dc * 16] = (unsigned short)pk_bf16(o[dc][r] * lr, 0.0f);
    }
}

// ---------------------------------------------------------------------------
// Output projection: out = attn_bf16 (4096x2048) @ Wpack^T. One wave per
// 64x64 supertile (4x4 WMMA tiles): 16 WMMAs per 16 b128 loads per k-step,
// zero conversion VALU (both operands pre-converted bf16).
// ---------------------------------------------------------------------------
__global__ __launch_bounds__(256) void proj_kernel(
    const unsigned short* __restrict__ attn, const unsigned short* __restrict__ w,
    float* __restrict__ out)
{
    const int lane = threadIdx.x & 31;
    const int col  = lane & 15;
    const int hlf  = lane >> 4;
    const int wid  = blockIdx.x * (blockDim.x >> 5) + (threadIdx.x >> 5);
    const int NT   = EDIM / 64;            // 32 column supertiles
    const int nt   = wid % NT;
    const int mt   = wid / NT;
    const int row  = mt * 64;
    const int jcol = nt * 64;

    v8f acc[4][4];
    #pragma unroll
    for (int mi = 0; mi < 4; ++mi)
        #pragma unroll
        for (int ni = 0; ni < 4; ++ni) acc[mi][ni] = v8f{};

    for (int kb = 0; kb < EDIM; kb += 32) {
        Frag af[4], bw[4];
        #pragma unroll
        for (int mi = 0; mi < 4; ++mi) {
            // A: row = attn row, K = 16*hlf + 2v + e (contiguous in e)
            const unsigned short* ap =
                attn + (size_t)(row + mi * 16 + col) * EDIM + kb + 16 * hlf;
            af[mi].q[0] = *(const uint4*)(ap);
            af[mi].q[1] = *(const uint4*)(ap + 8);
        }
        #pragma unroll
        for (int ni = 0; ni < 4; ++ni) {
            // B: lane col = j, K = 16*hlf + 2v + e (contiguous in e)
            const unsigned short* wp =
                w + (size_t)(jcol + ni * 16 + col) * EDIM + kb + 16 * hlf;
            bw[ni].q[0] = *(const uint4*)(wp);
            bw[ni].q[1] = *(const uint4*)(wp + 8);
        }
        #pragma unroll
        for (int mi = 0; mi < 4; ++mi)
            #pragma unroll
            for (int ni = 0; ni < 4; ++ni)
                acc[mi][ni] = __builtin_amdgcn_wmma_f32_16x16x32_bf16(
                    false, af[mi].bf, false, bw[ni].bf, (short)0, acc[mi][ni], false, false);
    }

    #pragma unroll
    for (int mi = 0; mi < 4; ++mi)
        #pragma unroll
        for (int ni = 0; ni < 4; ++ni)
            #pragma unroll
            for (int r = 0; r < 8; ++r)
                out[(size_t)(row + mi * 16 + r + hlf * 8) * EDIM + jcol + ni * 16 + col] =
                    acc[mi][ni][r];
}

extern "C" void kernel_launch(void* const* d_in, const int* in_sizes, int n_in,
                              void* d_out, int out_size, void* d_ws, size_t ws_size,
                              hipStream_t stream) {
    const float*         keys    = (const float*)d_in[0];
    const float*         values  = (const float*)d_in[1];
    const float*         queries = (const float*)d_in[2];
    const unsigned char* mask    = (const unsigned char*)d_in[3]; // bool[B,S]
    const float*         w_out   = (const float*)d_in[4];
    float*               out     = (float*)d_out;

    // workspace carve-up (56 MB total)
    char* ws = (char*)d_ws;
    unsigned short* Kp = (unsigned short*)(ws);                       // 16 MB
    unsigned short* Vp = (unsigned short*)(ws + (16ull << 20));       // 16 MB
    unsigned short* At = (unsigned short*)(ws + (32ull << 20));       // 16 MB
    unsigned short* Wp = (unsigned short*)(ws + (48ull << 20));       //  8 MB

    // bandwidth-bound bf16 pre-passes
    pack_k_kernel<<<(BATCH * HEADS * SEQ * (CDIM / 2)) / 256, 256, 0, stream>>>(
        keys, (unsigned*)Kp);
    pack_v_kernel<<<BATCH * HEADS * (SEQ / 32), 256, 0, stream>>>(values, Vp);
    pack_w_kernel<<<(EDIM * (EDIM / 2)) / 256, 256, 0, stream>>>(w_out, (unsigned*)Wp);

    // one 8-wave block per (b, head, 128-query tile)
    attn_fwd_kernel<<<BATCH * HEADS * (SEQ / 128), 256, 0, stream>>>(
        Kp, Vp, queries, mask, At);

    // (4096/64) * (2048/64) = 2048 waves, 8 waves per block
    const int tiles = (BATCH * SEQ / 64) * (EDIM / 64);
    proj_kernel<<<tiles / 8, 256, 0, stream>>>(At, Wp, out);
}
